// Mamba2_80977313399493
// MI455X (gfx1250) — matrix operand order repaired and checked by reading, not compiled, not run on hardware
//
#include <hip/hip_runtime.h>

// ---------------- problem constants ----------------
#define BATCH   2
#define SEQ     4096
#define HIDDEN  2048
#define HEADDIM 64
#define NSTATE  128
#define CONVK   4
#define CHUNK   64
#define INNER   4096
#define NHEADS  64
#define CONVCH  4352      // INNER + 2*NSTATE
#define DPROJ   8512      // 2*INNER + 2*NSTATE + NHEADS
#define NCHUNK  64        // SEQ / CHUNK

// ---------------- WMMA helpers (CDNA5 wave32) ----------------
typedef __attribute__((ext_vector_type(16))) __bf16 v16bf;
typedef __attribute__((ext_vector_type(8)))  float  v8f;

union BF16Frag { unsigned short us[16]; v16bf v; };

// Native hardware convert (v_cvt_*bf16*) instead of software round-to-even:
// one VALU op per element rather than bfe+add3+mov_b16.
__device__ __forceinline__ unsigned short f2bf(float f) {
  union { __bf16 b; unsigned short u; } cv;
  cv.b = (__bf16)f;
  return cv.u;
}

#define WMMA_BF16(a, b, c) \
  __builtin_amdgcn_wmma_f32_16x16x32_bf16(false, (a), false, (b), (short)0, (c), false, false)

// A operand: 16x32 bf16 from a bf16 LDS tile, row-major [m][k], leading dim ldm.
// This code's contiguous k order (same as frag_b): lanes 0-15 row m=lane, K={0..15}; lanes 16-31 row m=lane-16, K={16..31}.
__device__ __forceinline__ v16bf frag_a(const unsigned short* lds, int row0, int ldm,
                                        int k0, int lane) {
  BF16Frag r;
  const int l = lane & 15, hi = lane >> 4;
  const unsigned short* p = lds + (size_t)(row0 + l) * ldm + k0;
  const int cA = hi ? 16 : 0, cB = hi ? 24 : 8;
#pragma unroll
  for (int i = 0; i < 8; ++i) r.us[i] = p[cA + i];
#pragma unroll
  for (int i = 0; i < 8; ++i) r.us[8 + i] = p[cB + i];
  return r.v;
}

// B operand: 32x16 bf16; LDS tile holds B^T row-major [n][k], leading dim ldm.
__device__ __forceinline__ v16bf frag_b(const unsigned short* lds, int col0, int ldm,
                                        int k0, int lane) {
  BF16Frag r;
  const int l = lane & 15, hi = lane >> 4;
  const unsigned short* p = lds + (size_t)(col0 + l) * ldm + k0 + hi * 16;
#pragma unroll
  for (int i = 0; i < 16; ++i) r.us[i] = p[i];
  return r.v;
}

// Same fragment gathers but from raw f32 LDS tiles (async-copied), convert on read.
__device__ __forceinline__ v16bf frag_a_f32(const float* lds, int row0, int ldm,
                                            int k0, int lane) {
  BF16Frag r;
  const int l = lane & 15, hi = lane >> 4;
  const float* p = lds + (size_t)(row0 + l) * ldm + k0;
  const int cA = hi ? 16 : 0, cB = hi ? 24 : 8;
#pragma unroll
  for (int i = 0; i < 8; ++i) r.us[i] = f2bf(p[cA + i]);
#pragma unroll
  for (int i = 0; i < 8; ++i) r.us[8 + i] = f2bf(p[cB + i]);
  return r.v;
}

__device__ __forceinline__ v16bf frag_b_f32(const float* lds, int col0, int ldm,
                                            int k0, int lane) {
  BF16Frag r;
  const int l = lane & 15, hi = lane >> 4;
  const float* p = lds + (size_t)(col0 + l) * ldm + k0 + hi * 16;
#pragma unroll
  for (int i = 0; i < 16; ++i) r.us[i] = f2bf(p[i]);
  return r.v;
}

// ---------------- generic NT GEMM: C[M,N] = A[M,K] * B[N,K]^T ----------------
// Double-buffered K pipeline using CDNA5 async global->LDS copies (ASYNCcnt).
#define GBM 128
#define GBN 128
#define GBK 32
#define GLDF (GBK + 4)   // f32 LDS row stride (16B-aligned rows)
#define GPAD 8           // bf16 LDS pad for the small SSD kernels

// Issue one 128x32 f32 tile as 4 async b128 copies per thread-slice (8 per wave
// counting both A and B calls). Rows are clamped into [0, maxrow] so every wave
// always issues the same number of async ops (uniform s_wait_asynccnt bookkeeping);
// clamped rows only ever feed output columns that the store loop masks off.
__device__ __forceinline__ void issue_async_tile(const float* __restrict__ gbase, int ldg,
                                                 int maxrow, float* lds_base, int tid) {
#pragma unroll
  for (int it = 0; it < 4; ++it) {
    const int e = (it * 256 + tid) * 4;
    int r = e >> 5;
    const int c = e & 31;
    const int rr = (r < maxrow) ? r : maxrow;
    const float* g = gbase + (size_t)rr * ldg + c;
    const unsigned lds_off = (unsigned)(unsigned long long)(lds_base + r * GLDF + c);
    asm volatile("global_load_async_to_lds_b128 %0, %1, off"
                 :: "v"(lds_off), "v"(g)
                 : "memory");
  }
}

__global__ __launch_bounds__(256) void gemm_nt(const float* __restrict__ A,
                                               const float* __restrict__ B,
                                               float* __restrict__ C,
                                               int M, int N, int K) {
  __shared__ float Asf[2][GBM][GLDF];
  __shared__ float Bsf[2][GBN][GLDF];
  const int tid  = threadIdx.x;
  const int lane = tid & 31;
  const int wid  = tid >> 5;
  const int wm = wid >> 1, wn = wid & 1;   // 4x2 wave grid; wave tile 32x64
  const int bm = blockIdx.y * GBM;
  const int bn = blockIdx.x * GBN;
  const int bmaxrow = (N - bn) - 1;        // clamp for the N tail tile
  const v8f vz = {0.f, 0.f, 0.f, 0.f, 0.f, 0.f, 0.f, 0.f};
  v8f acc[2][4];
#pragma unroll
  for (int i = 0; i < 2; ++i)
#pragma unroll
    for (int j = 0; j < 4; ++j) acc[i][j] = vz;

  const int nk = K / GBK;
  // prologue: prefetch K-tile 0 into buffer 0
  issue_async_tile(A + (size_t)bm * K, K, GBM - 1, &Asf[0][0][0], tid);
  issue_async_tile(B + (size_t)bn * K, K, bmaxrow, &Bsf[0][0][0], tid);

  for (int ki = 0; ki < nk; ++ki) {
    const int cur = ki & 1;
    if (ki + 1 < nk) {
      const int k0 = (ki + 1) * GBK;
      issue_async_tile(A + (size_t)bm * K + k0, K, GBM - 1, &Asf[cur ^ 1][0][0], tid);
      issue_async_tile(B + (size_t)bn * K + k0, K, bmaxrow, &Bsf[cur ^ 1][0][0], tid);
      asm volatile("s_wait_asynccnt 8" ::: "memory");   // current tile's 8 copies done
    } else {
      asm volatile("s_wait_asynccnt 0" ::: "memory");
    }
    __syncthreads();  // all waves' async data for `cur` visible

    v16bf aF[2], bF[4];
#pragma unroll
    for (int tm = 0; tm < 2; ++tm)
      aF[tm] = frag_a_f32(&Asf[cur][0][0], wm * 32 + tm * 16, GLDF, 0, lane);
#pragma unroll
    for (int tn = 0; tn < 4; ++tn)
      bF[tn] = frag_b_f32(&Bsf[cur][0][0], wn * 64 + tn * 16, GLDF, 0, lane);
#pragma unroll
    for (int tm = 0; tm < 2; ++tm)
#pragma unroll
      for (int tn = 0; tn < 4; ++tn)
        acc[tm][tn] = WMMA_BF16(aF[tm], bF[tn], acc[tm][tn]);
    __syncthreads();  // readers done before this buffer is overwritten next round
  }

  const int mb = ((lane >> 4) << 3);
  const int nn = lane & 15;
#pragma unroll
  for (int tm = 0; tm < 2; ++tm)
#pragma unroll
    for (int tn = 0; tn < 4; ++tn) {
      const int n = bn + wn * 64 + tn * 16 + nn;
      if (n >= N) continue;
      const int m0 = bm + wm * 32 + tm * 16 + mb;
#pragma unroll
      for (int j = 0; j < 8; ++j)
        C[(size_t)(m0 + j) * N + n] = acc[tm][tn][j];
    }
}

// ---------------- depthwise conv (K=4, causal) + bias + SiLU ----------------
__global__ void conv_silu_kernel(const float* __restrict__ zx, const float* __restrict__ cw,
                                 const float* __restrict__ cb, float* __restrict__ xbc) {
  const size_t idx = (size_t)blockIdx.x * blockDim.x + threadIdx.x;
  const size_t total = (size_t)BATCH * SEQ * CONVCH;
  if (idx >= total) return;
  const int ch = (int)(idx % CONVCH);
  const int t  = (int)((idx / CONVCH) % SEQ);
  const int b  = (int)(idx / ((size_t)CONVCH * SEQ));
  float s = cb[ch];
#pragma unroll
  for (int j = 0; j < CONVK; ++j) {
    const int tt = t - (CONVK - 1) + j;
    if (tt >= 0)
      s += zx[((size_t)(b * SEQ + tt)) * DPROJ + INNER + ch] * cw[ch * CONVK + j];
  }
  s = s / (1.f + expf(-s));
  xbc[((size_t)(b * SEQ + t)) * CONVCH + ch] = s;
}

// ---------------- dt = softplus(raw + bias); dA = -exp(A_log)*dt ----------------
__global__ void dt_kernel(const float* __restrict__ zx, const float* __restrict__ dt_bias,
                          const float* __restrict__ A_log, float* __restrict__ dt,
                          float* __restrict__ dA) {
  const size_t idx = (size_t)blockIdx.x * blockDim.x + threadIdx.x;
  if (idx >= (size_t)BATCH * SEQ * NHEADS) return;
  const int h = (int)(idx % NHEADS);
  const float raw = zx[(idx / NHEADS) * DPROJ + (DPROJ - NHEADS) + h] + dt_bias[h];
  const float sp = (raw > 20.f) ? raw : log1pf(expf(raw));
  dt[idx] = sp;
  dA[idx] = -expf(A_log[h]) * sp;
}

// ---------------- conv_state output: [b, ch, k] = xBC_raw[b, T-4+k, ch] ----------------
__global__ void conv_state_kernel(const float* __restrict__ zx, float* __restrict__ cs) {
  const int idx = blockIdx.x * blockDim.x + threadIdx.x;
  if (idx >= BATCH * CONVCH * CONVK) return;
  const int k = idx % CONVK;
  const int ch = (idx / CONVK) % CONVCH;
  const int b = idx / (CONVK * CONVCH);
  cs[idx] = zx[((size_t)(b * SEQ + (SEQ - CONVK + k))) * DPROJ + INNER + ch];
}

// ---------------- per-chunk inclusive cumsum of dA -> acs[b,h,c,l] ----------------
__global__ void acs_kernel(const float* __restrict__ dA, float* __restrict__ acs) {
  const int idx = blockIdx.x * blockDim.x + threadIdx.x;
  if (idx >= BATCH * NHEADS * NCHUNK) return;
  const int c = idx & 63;
  const int h = (idx >> 6) & 63;
  const int b = idx >> 12;
  float run = 0.f;
  float* o = acs + (size_t)idx * CHUNK;
  for (int l = 0; l < CHUNK; ++l) {
    run += dA[((size_t)(b * SEQ + c * CHUNK + l)) * NHEADS + h];
    o[l] = run;
  }
}

// ---------------- G[b,c,l,s] = C_c[l,:] . B_c[s,:]  (K = NSTATE) ----------------
__global__ __launch_bounds__(128) void gmat_kernel(const float* __restrict__ xbc,
                                                   float* __restrict__ G) {
  __shared__ unsigned short Cs[CHUNK][NSTATE + GPAD];
  __shared__ unsigned short Bs[CHUNK][NSTATE + GPAD];
  const int c = blockIdx.x, b = blockIdx.y;
  const int tid = threadIdx.x, lane = tid & 31, wid = tid >> 5;
  const size_t rowbase = (size_t)(b * SEQ + c * CHUNK);
  for (int e = tid; e < CHUNK * NSTATE; e += 128) {
    const int l = e >> 7, n = e & 127;
    const float* src = xbc + (rowbase + l) * CONVCH + INNER;
    Bs[l][n] = f2bf(src[n]);
    Cs[l][n] = f2bf(src[NSTATE + n]);
  }
  __syncthreads();
  const int wm = wid >> 1, wn = wid & 1;  // 2x2 waves, 32x32 each
  const v8f vz = {0.f, 0.f, 0.f, 0.f, 0.f, 0.f, 0.f, 0.f};
  v8f acc[2][2];
#pragma unroll
  for (int i = 0; i < 2; ++i)
#pragma unroll
    for (int j = 0; j < 2; ++j) acc[i][j] = vz;
  for (int kk = 0; kk < NSTATE; kk += 32) {
    v16bf aF[2], bF[2];
#pragma unroll
    for (int tm = 0; tm < 2; ++tm)
      aF[tm] = frag_a(&Cs[0][0], wm * 32 + tm * 16, NSTATE + GPAD, kk, lane);
#pragma unroll
    for (int tn = 0; tn < 2; ++tn)
      bF[tn] = frag_b(&Bs[0][0], wn * 32 + tn * 16, NSTATE + GPAD, kk, lane);
#pragma unroll
    for (int tm = 0; tm < 2; ++tm)
#pragma unroll
      for (int tn = 0; tn < 2; ++tn)
        acc[tm][tn] = WMMA_BF16(aF[tm], bF[tn], acc[tm][tn]);
  }
  const int mb = ((lane >> 4) << 3), nn = lane & 15;
  float* g = G + ((size_t)(b * NCHUNK + c)) * CHUNK * CHUNK;
#pragma unroll
  for (int tm = 0; tm < 2; ++tm)
#pragma unroll
    for (int tn = 0; tn < 2; ++tn) {
      const int s = wn * 32 + tn * 16 + nn;
#pragma unroll
      for (int j = 0; j < 8; ++j) {
        const int l = wm * 32 + tm * 16 + mb + j;
        g[l * CHUNK + s] = acc[tm][tn][j];
      }
    }
}

// ---------------- per (b,c,h): Y_diag (masked G*L @ xdt) and chunk states ----------------
__global__ __launch_bounds__(128) void ydiag_states_kernel(
    const float* __restrict__ xbc, const float* __restrict__ dt,
    const float* __restrict__ acs, const float* __restrict__ G,
    float* __restrict__ Y, float* __restrict__ states) {
  __shared__ unsigned short sGL[CHUNK][CHUNK + GPAD];   // [l][s] = G*L (masked)
  __shared__ unsigned short sXT[CHUNK][CHUNK + GPAD];   // [p][l] = xdt^T
  __shared__ unsigned short sBT[NSTATE][CHUNK + GPAD];  // [n][l] = dec[l]*B[l][n]
  __shared__ float sAcs[CHUNK], sDt[CHUNK], sDec[CHUNK];
  const int h = blockIdx.x, c = blockIdx.y, b = blockIdx.z;
  const int tid = threadIdx.x, lane = tid & 31, w = tid >> 5;
  const size_t rowbase = (size_t)(b * SEQ + c * CHUNK);
  const size_t acsbase = ((size_t)(b * NHEADS + h) * NCHUNK + c) * CHUNK;

  if (tid < CHUNK) {
    sAcs[tid] = acs[acsbase + tid];
    sDt[tid] = dt[(rowbase + tid) * NHEADS + h];
  }
  __syncthreads();
  if (tid < CHUNK) sDec[tid] = expf(sAcs[CHUNK - 1] - sAcs[tid]);
  const float* g = G + ((size_t)(b * NCHUNK + c)) * CHUNK * CHUNK;
  for (int e = tid; e < CHUNK * CHUNK; e += 128) {
    const int l = e >> 6, q = e & 63;
    const float xv = xbc[(rowbase + l) * CONVCH + h * HEADDIM + q] * sDt[l];
    sXT[q][l] = f2bf(xv);
    float gl = 0.f;
    if (q <= l) gl = g[l * CHUNK + q] * expf(sAcs[l] - sAcs[q]);
    sGL[l][q] = f2bf(gl);
  }
  __syncthreads();
  for (int e = tid; e < CHUNK * NSTATE; e += 128) {
    const int l = e >> 7, n = e & 127;
    sBT[n][l] = f2bf(xbc[(rowbase + l) * CONVCH + INNER + n] * sDec[l]);
  }
  __syncthreads();

  const v8f vz = {0.f, 0.f, 0.f, 0.f, 0.f, 0.f, 0.f, 0.f};
  const int mb = ((lane >> 4) << 3), nn = lane & 15;

  // Y_diag[l][p] = sum_s GL[l][s] * xdt[s][p]   (K = CHUNK)
  {
    v8f acc[4] = {vz, vz, vz, vz};
    for (int kk = 0; kk < CHUNK; kk += 32) {
      const v16bf aF = frag_a(&sGL[0][0], w * 16, CHUNK + GPAD, kk, lane);
#pragma unroll
      for (int tn = 0; tn < 4; ++tn) {
        const v16bf bF = frag_b(&sXT[0][0], tn * 16, CHUNK + GPAD, kk, lane);
        acc[tn] = WMMA_BF16(aF, bF, acc[tn]);
      }
    }
#pragma unroll
    for (int tn = 0; tn < 4; ++tn) {
      const int p = tn * 16 + nn;
#pragma unroll
      for (int j = 0; j < 8; ++j) {
        const int l = w * 16 + mb + j;
        Y[((rowbase + l) * NHEADS + h) * HEADDIM + p] = acc[tn][j];
      }
    }
  }

  // states[p][n] = sum_l xdt[l][p] * dec[l]*B[l][n]   (K = CHUNK)
  {
    v8f acc[8] = {vz, vz, vz, vz, vz, vz, vz, vz};
    for (int kk = 0; kk < CHUNK; kk += 32) {
      const v16bf aF = frag_a(&sXT[0][0], w * 16, CHUNK + GPAD, kk, lane);
#pragma unroll
      for (int tn = 0; tn < 8; ++tn) {
        const v16bf bF = frag_b(&sBT[0][0], tn * 16, CHUNK + GPAD, kk, lane);
        acc[tn] = WMMA_BF16(aF, bF, acc[tn]);
      }
    }
    const size_t stbase =
        ((size_t)((b * NCHUNK + c) * NHEADS + h)) * HEADDIM * NSTATE;
#pragma unroll
    for (int tn = 0; tn < 8; ++tn) {
      const int n = tn * 16 + nn;
#pragma unroll
      for (int j = 0; j < 8; ++j) {
        const int p = w * 16 + mb + j;
        states[stbase + (size_t)p * NSTATE + n] = acc[tn][j];
      }
    }
  }
}

// ---------------- inter-chunk scan (in-place: states -> states_in) + final state ----------------
__global__ void scan_kernel(const float* __restrict__ acs, float* __restrict__ states,
                            float* __restrict__ ssm_out) {
  const size_t idx = (size_t)blockIdx.x * blockDim.x + threadIdx.x;
  if (idx >= (size_t)BATCH * NHEADS * HEADDIM * NSTATE) return;
  const int n = (int)(idx & (NSTATE - 1));
  const int p = (int)((idx >> 7) & 63);
  const int h = (int)((idx >> 13) & 63);
  const int b = (int)(idx >> 19);
  float R = 0.f;
  for (int c = 0; c < NCHUNK; ++c) {
    const size_t si =
        (((size_t)((b * NCHUNK + c) * NHEADS + h)) * HEADDIM + p) * NSTATE + n;
    const float s = states[si];
    const float dsum = acs[(((size_t)(b * NHEADS + h)) * NCHUNK + c) * CHUNK + CHUNK - 1];
    states[si] = R;  // state entering chunk c
    R = expf(dsum) * R + s;
  }
  ssm_out[(((size_t)(b * NHEADS + h)) * HEADDIM + p) * NSTATE + n] = R;
}

// ---------------- Y_off[l][p] += exp(Acs[l]) * sum_n C[l][n]*Sin[p][n] ----------------
__global__ __launch_bounds__(128) void yoff_kernel(const float* __restrict__ xbc,
                                                   const float* __restrict__ acs,
                                                   const float* __restrict__ states,
                                                   float* __restrict__ Y) {
  __shared__ unsigned short sC[CHUNK][NSTATE + GPAD];    // [l][n]
  __shared__ unsigned short sS[HEADDIM][NSTATE + GPAD];  // [p][n]
  __shared__ float sAcs[CHUNK];
  const int h = blockIdx.x, c = blockIdx.y, b = blockIdx.z;
  const int tid = threadIdx.x, lane = tid & 31, w = tid >> 5;
  const size_t rowbase = (size_t)(b * SEQ + c * CHUNK);
  const size_t acsbase = ((size_t)(b * NHEADS + h) * NCHUNK + c) * CHUNK;
  const size_t stbase = ((size_t)((b * NCHUNK + c) * NHEADS + h)) * HEADDIM * NSTATE;
  if (tid < CHUNK) sAcs[tid] = acs[acsbase + tid];
  for (int e = tid; e < CHUNK * NSTATE; e += 128) {
    const int l = e >> 7, n = e & 127;
    sC[l][n] = f2bf(xbc[(rowbase + l) * CONVCH + INNER + NSTATE + n]);
    sS[l][n] = f2bf(states[stbase + (size_t)l * NSTATE + n]);  // l plays role of p
  }
  __syncthreads();
  const v8f vz = {0.f, 0.f, 0.f, 0.f, 0.f, 0.f, 0.f, 0.f};
  v8f acc[4] = {vz, vz, vz, vz};
  for (int kk = 0; kk < NSTATE; kk += 32) {
    const v16bf aF = frag_a(&sC[0][0], w * 16, NSTATE + GPAD, kk, lane);
#pragma unroll
    for (int tn = 0; tn < 4; ++tn) {
      const v16bf bF = frag_b(&sS[0][0], tn * 16, NSTATE + GPAD, kk, lane);
      acc[tn] = WMMA_BF16(aF, bF, acc[tn]);
    }
  }
  const int mb = ((lane >> 4) << 3), nn = lane & 15;
#pragma unroll
  for (int tn = 0; tn < 4; ++tn) {
    const int p = tn * 16 + nn;
#pragma unroll
    for (int j = 0; j < 8; ++j) {
      const int l = w * 16 + mb + j;
      const float sdo = expf(sAcs[l]);
      const size_t yi = ((rowbase + l) * NHEADS + h) * HEADDIM + p;
      Y[yi] += sdo * acc[tn][j];
    }
  }
}

// ---------------- y = (Y + x*D) * silu(z); RMSNorm * norm_w (in place on Y) ----------------
__global__ __launch_bounds__(256) void gate_norm_kernel(const float* __restrict__ zx,
                                                        const float* __restrict__ xbc,
                                                        const float* __restrict__ Dp,
                                                        const float* __restrict__ nw,
                                                        float* __restrict__ Y) {
  __shared__ float red[256];
  const int bt = blockIdx.x;
  const int tid = threadIdx.x;
  const size_t zbase = (size_t)bt * DPROJ;
  const size_t xbase = (size_t)bt * CONVCH;
  const size_t ybase = (size_t)bt * INNER;
  float vals[INNER / 256];
  float ss = 0.f;
#pragma unroll
  for (int i = 0; i < INNER / 256; ++i) {
    const int e = i * 256 + tid;
    float y = Y[ybase + e] + xbc[xbase + e] * Dp[e >> 6];
    const float z = zx[zbase + e];
    y *= z / (1.f + expf(-z));
    vals[i] = y;
    ss += y * y;
  }
  red[tid] = ss;
  __syncthreads();
  for (int s = 128; s > 0; s >>= 1) {
    if (tid < s) red[tid] += red[tid + s];
    __syncthreads();
  }
  const float r = rsqrtf(red[0] / (float)INNER + 1e-5f);
#pragma unroll
  for (int i = 0; i < INNER / 256; ++i) {
    const int e = i * 256 + tid;
    Y[ybase + e] = vals[i] * r * nw[e];
  }
}

// ---------------- host launcher ----------------
extern "C" void kernel_launch(void* const* d_in, const int* in_sizes, int n_in,
                              void* d_out, int out_size, void* d_ws, size_t ws_size,
                              hipStream_t stream) {
  (void)in_sizes; (void)n_in; (void)out_size; (void)ws_size;
  const float* u      = (const float*)d_in[0];
  const float* W_in   = (const float*)d_in[1];
  const float* conv_w = (const float*)d_in[2];
  const float* conv_b = (const float*)d_in[3];
  const float* dtbias = (const float*)d_in[4];
  const float* A_log  = (const float*)d_in[5];
  const float* Dp     = (const float*)d_in[6];
  const float* nw     = (const float*)d_in[7];
  const float* W_out  = (const float*)d_in[8];
  float* out = (float*)d_out;

  float* ws = (float*)d_ws;
  size_t o = 0;
  float* zx  = ws + o; o += (size_t)BATCH * SEQ * DPROJ;                       // 69.7M
  float* xbc = ws + o; o += (size_t)BATCH * SEQ * CONVCH;                      // 35.7M
  float* dtb = ws + o; o += (size_t)BATCH * SEQ * NHEADS;                      // 0.5M
  float* dAb = ws + o; o += (size_t)BATCH * SEQ * NHEADS;                      // 0.5M
  float* acs = ws + o; o += (size_t)BATCH * NHEADS * NCHUNK * CHUNK;           // 0.5M
  float* G   = ws + o; o += (size_t)BATCH * NCHUNK * CHUNK * CHUNK;            // 0.5M
  float* st  = ws + o; o += (size_t)BATCH * NCHUNK * NHEADS * HEADDIM * NSTATE;// 67M
  float* Y   = ws + o;                                                         // 33.6M

  float* out_cs  = out + (size_t)BATCH * SEQ * HIDDEN;
  float* out_ssm = out_cs + (size_t)BATCH * CONVCH * CONVK;

  const int M = BATCH * SEQ;  // 8192

  // 1) in_proj: zxbcdt = u @ W_in^T
  gemm_nt<<<dim3((DPROJ + GBN - 1) / GBN, M / GBM), 256, 0, stream>>>(
      u, W_in, zx, M, DPROJ, HIDDEN);

  // 2) conv + SiLU, dt/dA, conv_state output
  {
    const size_t tot = (size_t)BATCH * SEQ * CONVCH;
    conv_silu_kernel<<<(unsigned)((tot + 255) / 256), 256, 0, stream>>>(
        zx, conv_w, conv_b, xbc);
  }
  dt_kernel<<<(unsigned)(((size_t)M * NHEADS + 255) / 256), 256, 0, stream>>>(
      zx, dtbias, A_log, dtb, dAb);
  conv_state_kernel<<<(BATCH * CONVCH * CONVK + 255) / 256, 256, 0, stream>>>(zx, out_cs);

  // 3) per-chunk cumsum
  acs_kernel<<<(BATCH * NHEADS * NCHUNK + 63) / 64, 64, 0, stream>>>(dAb, acs);

  // 4) G = C B^T per (b,c)
  gmat_kernel<<<dim3(NCHUNK, BATCH), 128, 0, stream>>>(xbc, G);

  // 5) Y_diag + chunk states per (b,c,h)
  ydiag_states_kernel<<<dim3(NHEADS, NCHUNK, BATCH), 128, 0, stream>>>(
      xbc, dtb, acs, G, Y, st);

  // 6) inter-chunk scan; writes states_in in place and final ssm_state to out
  scan_kernel<<<(unsigned)(((size_t)BATCH * NHEADS * HEADDIM * NSTATE + 255) / 256),
               256, 0, stream>>>(acs, st, out_ssm);

  // 7) Y += Y_off
  yoff_kernel<<<dim3(NHEADS, NCHUNK, BATCH), 128, 0, stream>>>(xbc, acs, st, Y);

  // 8) gate + RMSNorm (in place on Y)
  gate_norm_kernel<<<M, 256, 0, stream>>>(zx, xbc, Dp, nw, Y);

  // 9) out_proj: out = y @ W_out^T
  gemm_nt<<<dim3(HIDDEN / GBN, M / GBM), 256, 0, stream>>>(Y, W_out, out, M, HIDDEN, INNER);
}
